// CrossCityReprModel_48198122996220
// MI455X (gfx1250) — matrix operand order repaired and checked by reading, not compiled, not run on hardware
//
#include <hip/hip_runtime.h>
#include <hip/hip_bf16.h>

#define HD 128
#define RD 256

typedef __attribute__((ext_vector_type(16))) _Float16 v16h;
typedef __attribute__((ext_vector_type(8)))  float    v8f;

__device__ __forceinline__ float gelu_f(float x) {
  return 0.5f * x * (1.0f + erff(x * 0.70710678118654752440f));
}

// ------------------------------------------------------------------
// elementwise helpers
// ------------------------------------------------------------------
__global__ void fill_kernel(float* __restrict__ p, float v, long n) {
  long i = (long)blockIdx.x * blockDim.x + threadIdx.x;
  if (i < n) p[i] = v;
}
__global__ void copy_kernel(const float* __restrict__ a, float* __restrict__ o, long n) {
  long i = (long)blockIdx.x * blockDim.x + threadIdx.x;
  if (i < n) o[i] = a[i];
}
__global__ void negcopy_kernel(const float* __restrict__ a, float* __restrict__ o, long n) {
  long i = (long)blockIdx.x * blockDim.x + threadIdx.x;
  if (i < n) o[i] = -a[i];
}

// emb = concat(emb_lane[x0](32), emb_type[x1](32), emb_len[x2](64))
__global__ void embed_kernel(const int* __restrict__ x,
                             const float* __restrict__ el, const float* __restrict__ et,
                             const float* __restrict__ en, float* __restrict__ out, long n) {
  long i = (long)blockIdx.x * blockDim.x + threadIdx.x;
  if (i >= n) return;
  long node = i >> 7; int j = (int)(i & 127);
  float v;
  if (j < 32)       v = el[x[node * 3 + 0] * 32 + j];
  else if (j < 64)  v = et[x[node * 3 + 1] * 32 + (j - 32)];
  else              v = en[x[node * 3 + 2] * 64 + (j - 64)];
  out[i] = v;
}

// ------------------------------------------------------------------
// degree / dinv / spmm (edge-parallel, wave per edge)
// ------------------------------------------------------------------
__global__ void gcn_deg_kernel(const int* __restrict__ dst, float* __restrict__ deg, int nE) {
  int i = blockIdx.x * blockDim.x + threadIdx.x;
  if (i < nE) atomicAdd(&deg[dst[i]], 1.0f);
}
__global__ void cheb_deg_kernel(const int* __restrict__ src, const int* __restrict__ dst,
                                float* __restrict__ deg, int nE) {
  int i = blockIdx.x * blockDim.x + threadIdx.x;
  if (i < nE && src[i] != dst[i]) atomicAdd(&deg[src[i]], 1.0f);
}
__global__ void dinv_kernel(float* __restrict__ d, int n) {
  int i = blockIdx.x * blockDim.x + threadIdx.x;
  if (i < n) { float v = d[i]; d[i] = (v > 0.0f) ? rsqrtf(v) : 0.0f; }
}
// out = dinv[i]^2 * y + bias   (GCN self-loop + bias init)
__global__ void gcn_init_kernel(const float* __restrict__ Y, const float* __restrict__ dinv,
                                const float* __restrict__ bias, float* __restrict__ out, long n) {
  long i = (long)blockIdx.x * blockDim.x + threadIdx.x;
  if (i >= n) return;
  long node = i >> 7; int j = (int)(i & 127);
  float di = dinv[node];
  out[i] = di * di * Y[i] + bias[j];
}
// OUT[dst] += scale*dinv[s]*dinv[d] * Y[src]; cheb skips self edges
__global__ __launch_bounds__(256) void spmm_kernel(const int* __restrict__ src, const int* __restrict__ dst,
                                                   const float* __restrict__ dinv,
                                                   const float* __restrict__ Y, float* __restrict__ OUT,
                                                   int nE, float scale, int cheb) {
  int gid = blockIdx.x * blockDim.x + threadIdx.x;
  int e = gid >> 5;
  if (e >= nE) return;
  int lane = gid & 31;
  int s = src[e], d = dst[e];
  if (cheb && s == d) return;
  float w = dinv[s] * dinv[d] * scale;
  float4 v = ((const float4*)(Y + (long)s * HD))[lane];
  float* o = OUT + (long)d * HD + lane * 4;
  atomicAdd(o + 0, w * v.x); atomicAdd(o + 1, w * v.y);
  atomicAdd(o + 2, w * v.z); atomicAdd(o + 3, w * v.w);
}

// ------------------------------------------------------------------
// layernorm (wave per row of 128) with optional gelu before/after
// ------------------------------------------------------------------
#define LN_PREGELU 1
#define LN_POSTGELU 2
__global__ __launch_bounds__(256) void ln_kernel(const float* __restrict__ X,
                                                 const float* __restrict__ g, const float* __restrict__ b,
                                                 float* __restrict__ OUT, int ldo, int nrows, int flags) {
  int row = blockIdx.x * 8 + (threadIdx.x >> 5);
  if (row >= nrows) return;
  int lane = threadIdx.x & 31;
  float4 v = ((const float4*)(X + (long)row * HD))[lane];
  if (flags & LN_PREGELU) { v.x = gelu_f(v.x); v.y = gelu_f(v.y); v.z = gelu_f(v.z); v.w = gelu_f(v.w); }
  float s = v.x + v.y + v.z + v.w;
  for (int m = 16; m; m >>= 1) s += __shfl_xor(s, m, 32);
  float mean = s * (1.0f / 128.0f);
  float dx = v.x - mean, dy = v.y - mean, dz = v.z - mean, dw = v.w - mean;
  float ss = dx * dx + dy * dy + dz * dz + dw * dw;
  for (int m = 16; m; m >>= 1) ss += __shfl_xor(ss, m, 32);
  float r = rsqrtf(ss * (1.0f / 128.0f) + 1e-5f);
  float* o = OUT + (long)row * ldo + lane * 4;
  int j = lane * 4;
  float o0 = dx * r * g[j + 0] + b[j + 0];
  float o1 = dy * r * g[j + 1] + b[j + 1];
  float o2 = dz * r * g[j + 2] + b[j + 2];
  float o3 = dw * r * g[j + 3] + b[j + 3];
  if (flags & LN_POSTGELU) { o0 = gelu_f(o0); o1 = gelu_f(o1); o2 = gelu_f(o2); o3 = gelu_f(o3); }
  o[0] = o0; o[1] = o1; o[2] = o2; o[3] = o3;
}

// ------------------------------------------------------------------
// generic WMMA GEMM: OUT[M x 128] = alpha * X[M x K] @ W[K x 128] (+bias)(gelu)
// flags: gelu | accumulate | X transposed (X[k*ldx+m]) | W transposed (W[n*ldw+k]) | atomic split-K
// ------------------------------------------------------------------
#define GF_GELU 1
#define GF_ACC 2
#define GF_XT 4
#define GF_WT 8
#define GF_ATOMIC 16

__global__ __launch_bounds__(256) void gemm_wmma(const float* __restrict__ X, int ldx,
                                                 const float* __restrict__ W, int ldw,
                                                 const float* __restrict__ bias,
                                                 float* __restrict__ OUT, int ldo,
                                                 int M, int K, float alpha, int flags, int kchunk) {
  __shared__ _Float16 As[64][132];
  __shared__ _Float16 Bs[128][132];
  int tid = threadIdx.x;
  int wave = tid >> 5, lane = tid & 31;
  int m0 = blockIdx.x * 64;
  int rt = wave >> 1;           // row tile 0..3 (16 rows each)
  int cb = (wave & 1) * 4;      // col tile base (4 tiles of 16 cols each)
  int kstart = (flags & GF_ATOMIC) ? blockIdx.y * kchunk : 0;
  int kend   = (flags & GF_ATOMIC) ? min(K, kstart + kchunk) : K;

  v8f acc[4];
  v8f zero = {0.f, 0.f, 0.f, 0.f, 0.f, 0.f, 0.f, 0.f};
  for (int t = 0; t < 4; t++) acc[t] = zero;

  for (int kk = kstart; kk < kend; kk += 128) {
    __syncthreads();
    if (flags & GF_XT) {
      for (int i = tid; i < 64 * 128; i += 256) {
        int k = i >> 6, m = i & 63;
        int gm = m0 + m, gk = kk + k;
        float v = (gm < M && gk < kend) ? X[(long)gk * ldx + gm] : 0.0f;
        As[m][k] = (_Float16)v;
      }
    } else {
      for (int i = tid; i < 64 * 128; i += 256) {
        int m = i >> 7, k = i & 127;
        int gm = m0 + m, gk = kk + k;
        float v = (gm < M && gk < kend) ? X[(long)gm * ldx + gk] : 0.0f;
        As[m][k] = (_Float16)v;
      }
    }
    if (flags & GF_WT) {
      for (int i = tid; i < 128 * 128; i += 256) {
        int n = i >> 7, k = i & 127;
        int gk = kk + k;
        float v = (gk < kend) ? W[(long)n * ldw + gk] : 0.0f;
        Bs[n][k] = (_Float16)v;
      }
    } else {
      for (int i = tid; i < 128 * 128; i += 256) {
        int k = i >> 7, n = i & 127;
        int gk = kk + k;
        float v = (gk < kend) ? W[(long)gk * ldw + n] : 0.0f;
        Bs[n][k] = (_Float16)v;
      }
    }
    __syncthreads();

    int row = rt * 16 + (lane & 15);
    int kbA = (lane >> 4) * 8;
    int kbB = (lane >> 4) * 16;
    int ncl = lane & 15;
    for (int kc = 0; kc < 4; kc++) {
      v16h a;
#pragma unroll
      for (int j = 0; j < 8; j++) {
        a[j]     = As[row][kc * 32 + kbA + j];
        a[8 + j] = As[row][kc * 32 + 16 + kbA + j];
      }
#pragma unroll
      for (int t = 0; t < 4; t++) {
        int n = (cb + t) * 16 + ncl;
        v16h bfr;
#pragma unroll
        for (int j = 0; j < 8; j++) { bfr[j] = Bs[n][kc * 32 + kbA + j]; bfr[8 + j] = Bs[n][kc * 32 + 16 + kbA + j]; }
        acc[t] = __builtin_amdgcn_wmma_f32_16x16x32_f16(false, a, false, bfr,
                                                        (short)0, acc[t], false, false);
      }
    }
  }

  int half = lane >> 4, ncl = lane & 15;
  for (int t = 0; t < 4; t++) {
    int n = (cb + t) * 16 + ncl;
#pragma unroll
    for (int v = 0; v < 8; v++) {
      int m = m0 + rt * 16 + half * 8 + v;
      if (m < M) {
        float val = acc[t][v] * alpha;
        if (bias) val += bias[n];
        if (flags & GF_GELU) val = gelu_f(val);
        long off = (long)m * ldo + n;
        if (flags & GF_ATOMIC) atomicAdd(&OUT[off], val);
        else if (flags & GF_ACC) OUT[off] += val;
        else OUT[off] = val;
      }
    }
  }
}

// ------------------------------------------------------------------
// softmax over nodes (axis=1 of R x N), assign stored node-major [n*256+r]
// ------------------------------------------------------------------
__global__ __launch_bounds__(256) void smax_partial_kernel(const float* __restrict__ A,
                                                           float* __restrict__ Pm, float* __restrict__ Ps,
                                                           int n, int chunk) {
  int r = threadIdx.x, b = blockIdx.x;
  int n0 = b * chunk, n1 = min(n, n0 + chunk);
  float m = -3.0e38f, s = 0.0f;
  for (int i = n0; i < n1; i++) {
    float x = A[(long)i * RD + r];
    if (x > m) { s = s * __expf(m - x) + 1.0f; m = x; }
    else s += __expf(x - m);
  }
  Pm[b * RD + r] = m; Ps[b * RD + r] = s;
}
__global__ void smax_combine_kernel(const float* __restrict__ Pm, const float* __restrict__ Ps,
                                    float* __restrict__ Rm, float* __restrict__ Rs, int nb) {
  int r = threadIdx.x;
  float m = -3.0e38f, s = 0.0f;
  for (int b = 0; b < nb; b++) {
    float pm = Pm[b * RD + r], ps = Ps[b * RD + r];
    if (pm > m) { s = s * __expf(m - pm) + ps; m = pm; }
    else s += ps * __expf(pm - m);
  }
  Rm[r] = m; Rs[r] = s;
}
__global__ void smax_norm_kernel(float* __restrict__ A, const float* __restrict__ Rm,
                                 const float* __restrict__ Rs, long total) {
  long i = (long)blockIdx.x * blockDim.x + threadIdx.x;
  if (i >= total) return;
  int r = (int)(i & 255);
  A[i] = __expf(A[i] - Rm[r]) / Rs[r];
}

// seg2reg = argmax over r (wave per node)
__global__ __launch_bounds__(256) void argmax_kernel(const float* __restrict__ A, int* __restrict__ SR, int n) {
  int node = blockIdx.x * 8 + (threadIdx.x >> 5);
  if (node >= n) return;
  int lane = threadIdx.x & 31;
  float best = -3.0e38f; int bi = 0;
  for (int j = lane; j < RD; j += 32) {
    float x = A[(long)node * RD + j];
    if (x > best) { best = x; bi = j; }
  }
  for (int m = 16; m; m >>= 1) {
    float ob = __shfl_xor(best, m, 32);
    int oi = __shfl_xor(bi, m, 32);
    if (ob > best || (ob == best && oi < bi)) { best = ob; bi = oi; }
  }
  if (lane == 0) SR[node] = bi;
}

__global__ void hist_kernel(const int* __restrict__ src, const int* __restrict__ dst,
                            const int* __restrict__ SR, float* __restrict__ A, int nE) {
  int i = blockIdx.x * blockDim.x + threadIdx.x;
  if (i < nE) atomicAdd(&A[SR[src[i]] * RD + SR[dst[i]]], 1.0f);
}
__global__ __launch_bounds__(256) void wmat_kernel(const float* __restrict__ A, float* __restrict__ Wm) {
  __shared__ float red[256];
  int r = blockIdx.x, t = threadIdx.x;
  float a = A[r * RD + t];
  red[t] = a; __syncthreads();
  for (int s = 128; s; s >>= 1) { if (t < s) red[t] += red[t + s]; __syncthreads(); }
  Wm[r * RD + t] = a / fmaxf(red[0], 1e-6f);
}

// ------------------------------------------------------------------
// region transformer attention (block per (t,h), threads = s over R)
// ------------------------------------------------------------------
__global__ __launch_bounds__(256) void tattn_kernel(const float* __restrict__ Q, const float* __restrict__ Kf,
                                                    const float* __restrict__ V, const float* __restrict__ We,
                                                    const float* __restrict__ Wm, const float* __restrict__ A,
                                                    float* __restrict__ OUT) {
  int t = blockIdx.x, h = blockIdx.y, s = threadIdx.x;
  __shared__ float qsh[32], wesh[32], red[256], alsh[256];
  if (s < 32) { qsh[s] = Q[t * HD + h * 32 + s]; wesh[s] = We[h * 32 + s]; }
  __syncthreads();
  float qe = 0.0f, qk = 0.0f;
  for (int dd = 0; dd < 32; dd++) {
    float qv = qsh[dd];
    qe += qv * wesh[dd];
    qk += qv * Kf[s * HD + h * 32 + dd];
  }
  float wst = Wm[s * RD + t];
  float score = (qk + wst * qe) * 0.17677669529663687f; // 1/sqrt(32)
  bool msk = A[s * RD + t] > 0.0f;
  red[s] = msk ? score : -1e30f; __syncthreads();
  for (int o = 128; o; o >>= 1) { if (s < o) red[s] = fmaxf(red[s], red[s + o]); __syncthreads(); }
  float smax = red[0]; __syncthreads();
  float ex = msk ? __expf(score - smax) : 0.0f;
  red[s] = ex; __syncthreads();
  for (int o = 128; o; o >>= 1) { if (s < o) red[s] += red[s + o]; __syncthreads(); }
  float tot = fmaxf(red[0], 1e-16f); __syncthreads();
  float alpha = ex / tot;
  alsh[s] = alpha;
  red[s] = alpha * wst; __syncthreads();
  for (int o = 128; o; o >>= 1) { if (s < o) red[s] += red[s + o]; __syncthreads(); }
  float aw = red[0]; __syncthreads();
  if (s < 32) {
    float acc = 0.0f;
    for (int ss = 0; ss < RD; ss++) acc += alsh[ss] * V[ss * HD + h * 32 + s];
    OUT[t * HD + h * 32 + s] = acc + aw * wesh[s];
  }
}

// ------------------------------------------------------------------
// misc small kernels
// ------------------------------------------------------------------
__global__ void centers_adapt_kernel(const float* __restrict__ centers, const float* __restrict__ city_emb,
                                     const int* __restrict__ city, const float* __restrict__ adapt_W,
                                     float* __restrict__ CA) {
  int i = blockIdx.x * blockDim.x + threadIdx.x;
  if (i >= RD * HD) return;
  int c = city[0];
  float acc = centers[i];
  for (int r = 0; r < 8; r++) acc += city_emb[c * 8 + r] * adapt_W[(long)r * (RD * HD) + i];
  CA[i] = acc;
}
__global__ void film_gb_kernel(const float* __restrict__ city_emb, const int* __restrict__ city,
                               const float* __restrict__ film_W, float* __restrict__ GB) {
  int j = threadIdx.x;  // 0..255
  int c = city[0];
  float acc = 0.0f;
  for (int r = 0; r < 8; r++) acc += city_emb[c * 8 + r] * film_W[r * 256 + j];
  GB[j] = acc;
}
// resid = (init - slr)*(1+gb[:H]) + gb[H:]   (in place on init buffer)
__global__ void resid_film_kernel(float* __restrict__ R_, const float* __restrict__ SLR,
                                  const float* __restrict__ GB, long n) {
  long i = (long)blockIdx.x * blockDim.x + threadIdx.x;
  if (i >= n) return;
  int j = (int)(i & 127);
  R_[i] = (R_[i] - SLR[i]) * (1.0f + GB[j]) + GB[128 + j];
}
// gate[n] = sigmoid(dot(t1[n], g2_W) + g2_b)
__global__ __launch_bounds__(256) void gate_kernel(const float* __restrict__ T1, const float* __restrict__ g2W,
                                                   const float* __restrict__ g2b, float* __restrict__ G, int n) {
  int node = blockIdx.x * 8 + (threadIdx.x >> 5);
  if (node >= n) return;
  int lane = threadIdx.x & 31;
  float4 v = ((const float4*)(T1 + (long)node * HD))[lane];
  float4 w = ((const float4*)g2W)[lane];
  float s = v.x * w.x + v.y * w.y + v.z * w.z + v.w * w.w;
  for (int m = 16; m; m >>= 1) s += __shfl_xor(s, m, 32);
  if (lane == 0) G[node] = 1.0f / (1.0f + __expf(-(s + g2b[0])));
}
__global__ void fuse_kernel(const float* __restrict__ Wcat, const float* __restrict__ G,
                            float* __restrict__ OUT, long n) {
  long i = (long)blockIdx.x * blockDim.x + threadIdx.x;
  if (i >= n) return;
  long node = i >> 7; int j = (int)(i & 127);
  float g = G[node];
  OUT[i] = g * Wcat[node * 256 + j] + (1.0f - g) * Wcat[node * 256 + 128 + j];
}

// ------------------------------------------------------------------
static inline void launch_gemm(hipStream_t st, const float* X, int ldx, const float* W, int ldw,
                               const float* bias, float* OUT, int ldo, int M, int K,
                               float alpha, int flags, int kchunk = 0, int kyb = 1) {
  dim3 grid((M + 63) / 64, kyb);
  gemm_wmma<<<grid, 256, 0, st>>>(X, ldx, W, ldw, bias, OUT, ldo, M, K, alpha, flags, kchunk);
}

extern "C" void kernel_launch(void* const* d_in, const int* in_sizes, int n_in,
                              void* d_out, int out_size, void* d_ws, size_t ws_size,
                              hipStream_t stream) {
  (void)n_in; (void)out_size; (void)ws_size;
  const int NN = in_sizes[0] / 3;
  const int NE = in_sizes[1] / 2;
  const int* xin = (const int*)d_in[0];
  const int* src = (const int*)d_in[1];
  const int* dst = src + NE;
  const int* city = (const int*)d_in[2];
  // params dict flattened in sorted-key order after the 3 positional inputs
#define P(i) ((const float*)d_in[3 + (i)])
  const float *adapt_W = P(0), *c1e_W = P(1), *c1k_W = P(2), *c1k_b = P(3), *c1q_W = P(4), *c1q_b = P(5),
              *c1s_W = P(6), *c1s_b = P(7), *c1v_W = P(8), *c1v_b = P(9), *c2e_W = P(10), *c2k_W = P(11),
              *c2k_b = P(12), *c2q_W = P(13), *c2q_b = P(14), *c2s_W = P(15), *c2s_b = P(16), *c2v_W = P(17),
              *c2v_b = P(18), *capk_W = P(19), *capk_b = P(20), *capq_W = P(21), *capq_b = P(22),
              *centers = P(23), *cheb_hi = P(24), *cheb_hi_b = P(25), *cheb_lo = P(26), *cheb_lo_b = P(27),
              *city_emb = P(28), *emb_lane = P(29), *emb_len = P(30), *emb_type = P(31), *film_W = P(32),
              *g1_W = P(33), *g1_b = P(34), *g2_W = P(35), *g2_b = P(36), *gcn1_W = P(37), *gcn1_b = P(38),
              *gcn2_W = P(39), *gcn2_b = P(40), *ln_hi_b = P(41), *ln_hi_g = P(42), *ln_lo_b = P(43),
              *ln_lo_g = P(44), *ln_r1_b = P(45), *ln_r1_g = P(46), *ln_r2_b = P(47), *ln_r2_g = P(48),
              *ln_s1_b = P(49), *ln_s1_g = P(50), *ln_s2_b = P(51), *ln_s2_g = P(52), *lnp_b = P(53),
              *lnp_g = P(54), *o1_W = P(55), *o1_b = P(56), *o2_W = P(57), *o2_b = P(58), *p1_W = P(59),
              *p1_b = P(60), *p2_W = P(61), *p2_b = P(62), *proj_W = P(63), *proj_b = P(64);
#undef P

  // workspace layout
  float* ws = (float*)d_ws;
  const size_t NH = (size_t)NN * HD;
  float* W0 = ws;
  float* W1 = W0 + NH;
  float* W2 = W1 + NH;
  float* W3 = W2 + NH;
  float* W4 = W3 + NH;
  float* Wcat = W4 + NH;                 // 2*NH, hcat = [low | high]
  float* A0 = Wcat + 2 * NH;             // assign, node-major N x 256
  float* DG = A0 + (size_t)NN * RD;      // gcn dinv
  float* DC = DG + NN;                   // cheb dinv
  float* Gt = DC + NN;                   // gate
  int* SR = (int*)(Gt + NN);             // seg2reg
  float* Pm = (float*)(SR + NN);
  float* Ps = Pm + 256 * 256;
  float* Rm = Ps + 256 * 256;
  float* Rs = Rm + 256;
  float* GB = Rs + 256;
  float* CA = GB + 256;
  float* Qc = CA + RD * HD;
  float* RF = Qc + RD * HD;
  float* TQ = RF + RD * HD;
  float* TK = TQ + RD * HD;
  float* TV = TK + RD * HD;
  float* TO = TV + RD * HD;
  float* RH = TO + RD * HD;
  float* PROJ = RH + RD * HD;
  float* Amat = PROJ + RD * HD;
  float* Wm = Amat + RD * RD;

  const long NHL = (long)NN * HD;
  const int NBn = (int)((NHL + 255) / 256);
  const int NBe = (NE + 255) / 256;
  const int NBs = (int)(((long)NE * 32 + 255) / 256);
  const int NB8 = (NN + 7) / 8;

  // ---- embeddings + projection ----
  embed_kernel<<<NBn, 256, 0, stream>>>(xin, emb_lane, emb_type, emb_len, W0, NHL);
  launch_gemm(stream, W0, 128, proj_W, 128, proj_b, W1, 128, NN, 128, 1.0f, 0);   // init_feat=W1

  // ---- graph normalizations ----
  fill_kernel<<<(NN + 255) / 256, 256, 0, stream>>>(DG, 1.0f, NN);
  gcn_deg_kernel<<<NBe, 256, 0, stream>>>(dst, DG, NE);
  dinv_kernel<<<(NN + 255) / 256, 256, 0, stream>>>(DG, NN);
  fill_kernel<<<(NN + 255) / 256, 256, 0, stream>>>(DC, 0.0f, NN);
  cheb_deg_kernel<<<NBe, 256, 0, stream>>>(src, dst, DC, NE);
  dinv_kernel<<<(NN + 255) / 256, 256, 0, stream>>>(DC, NN);

  // ---- GCN 1 ----
  launch_gemm(stream, W1, 128, gcn1_W, 128, nullptr, W2, 128, NN, 128, 1.0f, 0);
  gcn_init_kernel<<<NBn, 256, 0, stream>>>(W2, DG, gcn1_b, W3, NHL);
  spmm_kernel<<<NBs, 256, 0, stream>>>(src, dst, DG, W2, W3, NE, 1.0f, 0);
  ln_kernel<<<NB8, 256, 0, stream>>>(W3, ln_s1_g, ln_s1_b, W3, 128, NN, LN_POSTGELU);
  // ---- GCN 2 ----
  launch_gemm(stream, W3, 128, gcn2_W, 128, nullptr, W2, 128, NN, 128, 1.0f, 0);
  gcn_init_kernel<<<NBn, 256, 0, stream>>>(W2, DG, gcn2_b, W4, NHL);
  spmm_kernel<<<NBs, 256, 0, stream>>>(src, dst, DG, W2, W4, NE, 1.0f, 0);
  ln_kernel<<<NB8, 256, 0, stream>>>(W4, ln_s2_g, ln_s2_b, W4, 128, NN, LN_POSTGELU); // seg_feat=W4

  // ---- city adaptation ----
  centers_adapt_kernel<<<(RD * HD + 255) / 256, 256, 0, stream>>>(centers, city_emb, city, adapt_W, CA);
  film_gb_kernel<<<1, 256, 0, stream>>>(city_emb, city, film_W, GB);

  // ---- capsule assignment ----
  launch_gemm(stream, CA, 128, capq_W, 128, capq_b, Qc, 128, RD, 128, 1.0f, 0);    // q
  launch_gemm(stream, W4, 128, capk_W, 128, capk_b, W2, 128, NN, 128, 1.0f, 0);    // k=W2
  // logits node-major: A0[n][r] = k.q / sqrt(H)  (two 128-col halves via W-transposed mode)
  launch_gemm(stream, W2, 128, Qc, 128, nullptr, A0, 256, NN, 128, 0.08838834764831845f, GF_WT);
  launch_gemm(stream, W2, 128, Qc + 128 * 128, 128, nullptr, A0 + 128, 256, NN, 128, 0.08838834764831845f, GF_WT);
  // softmax over nodes per region
  int chunk = (NN + 255) / 256;
  smax_partial_kernel<<<256, 256, 0, stream>>>(A0, Pm, Ps, NN, chunk);
  smax_combine_kernel<<<1, 256, 0, stream>>>(Pm, Ps, Rm, Rs, 256);
  smax_norm_kernel<<<(int)(((long)NN * RD + 255) / 256), 256, 0, stream>>>(A0, Rm, Rs, (long)NN * RD);
  argmax_kernel<<<NB8, 256, 0, stream>>>(A0, SR, NN);
  // region_feat = assign @ seg_feat + centers  (atomic split-K WMMA)
  copy_kernel<<<(RD * HD + 255) / 256, 256, 0, stream>>>(CA, RF, RD * HD);
  {
    int kc = 8192, kyb = (NN + kc - 1) / kc;
    launch_gemm(stream, A0, 256, W4, 128, nullptr, RF, 128, RD, NN, 1.0f, GF_XT | GF_ATOMIC, kc, kyb);
  }

  // ---- region adjacency ----
  fill_kernel<<<(RD * RD + 255) / 256, 256, 0, stream>>>(Amat, 0.0f, RD * RD);
  hist_kernel<<<NBe, 256, 0, stream>>>(src, dst, SR, Amat, NE);
  wmat_kernel<<<RD, 256, 0, stream>>>(Amat, Wm);

  // ---- tconv c1 ----
  launch_gemm(stream, RF, 128, c1q_W, 128, c1q_b, TQ, 128, RD, 128, 1.0f, 0);
  launch_gemm(stream, RF, 128, c1k_W, 128, c1k_b, TK, 128, RD, 128, 1.0f, 0);
  launch_gemm(stream, RF, 128, c1v_W, 128, c1v_b, TV, 128, RD, 128, 1.0f, 0);
  tattn_kernel<<<dim3(RD, 4), 256, 0, stream>>>(TQ, TK, TV, c1e_W, Wm, Amat, TO);
  launch_gemm(stream, RF, 128, c1s_W, 128, c1s_b, TO, 128, RD, 128, 1.0f, GF_ACC);
  ln_kernel<<<(RD + 7) / 8, 256, 0, stream>>>(TO, ln_r1_g, ln_r1_b, RH, 128, RD, LN_POSTGELU);
  // ---- tconv c2 ----
  launch_gemm(stream, RH, 128, c2q_W, 128, c2q_b, TQ, 128, RD, 128, 1.0f, 0);
  launch_gemm(stream, RH, 128, c2k_W, 128, c2k_b, TK, 128, RD, 128, 1.0f, 0);
  launch_gemm(stream, RH, 128, c2v_W, 128, c2v_b, TV, 128, RD, 128, 1.0f, 0);
  tattn_kernel<<<dim3(RD, 4), 256, 0, stream>>>(TQ, TK, TV, c2e_W, Wm, Amat, TO);
  launch_gemm(stream, RH, 128, c2s_W, 128, c2s_b, TO, 128, RD, 128, 1.0f, GF_ACC);
  ln_kernel<<<(RD + 7) / 8, 256, 0, stream>>>(TO, ln_r2_g, ln_r2_b, RF, 128, RD, LN_POSTGELU);
  // ---- projection head on regions ----
  launch_gemm(stream, RF, 128, p1_W, 128, p1_b, TQ, 128, RD, 128, 1.0f, GF_GELU);
  launch_gemm(stream, TQ, 128, p2_W, 128, p2_b, TK, 128, RD, 128, 1.0f, 0);
  ln_kernel<<<(RD + 7) / 8, 256, 0, stream>>>(TK, lnp_g, lnp_b, PROJ, 128, RD, 0);

  // ---- seg_low_raw = assign^T @ proj_reg ----
  launch_gemm(stream, A0, 256, PROJ, 128, nullptr, W3, 128, NN, 256, 1.0f, 0);     // W3 = seg_low_raw
  // resid (in place on W1) with FiLM
  resid_film_kernel<<<NBn, 256, 0, stream>>>(W1, W3, GB, NHL);

  // ---- Chebyshev low-pass (K=5), x = W3, acc = W4 ----
  launch_gemm(stream, W3, 128, cheb_lo, 128, cheb_lo_b, W4, 128, NN, 128, 1.0f, 0);
  fill_kernel<<<NBn, 256, 0, stream>>>(W0, 0.0f, NHL);
  spmm_kernel<<<NBs, 256, 0, stream>>>(src, dst, DC, W3, W0, NE, -1.0f, 1);        // Tx1=W0
  launch_gemm(stream, W0, 128, cheb_lo + 16384, 128, nullptr, W4, 128, NN, 128, 1.0f, GF_ACC);
  negcopy_kernel<<<NBn, 256, 0, stream>>>(W3, W2, NHL);                            // Tx2 = 2*L*Tx1 - Tx0
  spmm_kernel<<<NBs, 256, 0, stream>>>(src, dst, DC, W0, W2, NE, -2.0f, 1);
  launch_gemm(stream, W2, 128, cheb_lo + 2 * 16384, 128, nullptr, W4, 128, NN, 128, 1.0f, GF_ACC);
  negcopy_kernel<<<NBn, 256, 0, stream>>>(W0, W3, NHL);                            // rotate: Tx0=W0,Tx1=W2
  spmm_kernel<<<NBs, 256, 0, stream>>>(src, dst, DC, W2, W3, NE, -2.0f, 1);
  launch_gemm(stream, W3, 128, cheb_lo + 3 * 16384, 128, nullptr, W4, 128, NN, 128, 1.0f, GF_ACC);
  negcopy_kernel<<<NBn, 256, 0, stream>>>(W2, W0, NHL);                            // Tx0=W2,Tx1=W3
  spmm_kernel<<<NBs, 256, 0, stream>>>(src, dst, DC, W3, W0, NE, -2.0f, 1);
  launch_gemm(stream, W0, 128, cheb_lo + 4 * 16384, 128, nullptr, W4, 128, NN, 128, 1.0f, GF_ACC);
  ln_kernel<<<NB8, 256, 0, stream>>>(W4, ln_lo_g, ln_lo_b, Wcat, 256, NN, LN_PREGELU);       // low -> hcat[:, :128]

  // ---- Chebyshev high-pass (K=3), x = resid (W1), acc = W2 ----
  launch_gemm(stream, W1, 128, cheb_hi, 128, cheb_hi_b, W2, 128, NN, 128, 1.0f, 0);
  fill_kernel<<<NBn, 256, 0, stream>>>(W0, 0.0f, NHL);
  spmm_kernel<<<NBs, 256, 0, stream>>>(src, dst, DC, W1, W0, NE, -1.0f, 1);
  launch_gemm(stream, W0, 128, cheb_hi + 16384, 128, nullptr, W2, 128, NN, 128, 1.0f, GF_ACC);
  negcopy_kernel<<<NBn, 256, 0, stream>>>(W1, W3, NHL);
  spmm_kernel<<<NBs, 256, 0, stream>>>(src, dst, DC, W0, W3, NE, -2.0f, 1);
  launch_gemm(stream, W3, 128, cheb_hi + 2 * 16384, 128, nullptr, W2, 128, NN, 128, 1.0f, GF_ACC);
  ln_kernel<<<NB8, 256, 0, stream>>>(W2, ln_hi_g, ln_hi_b, Wcat + 128, 256, NN, LN_PREGELU); // high -> hcat[:, 128:]

  // ---- gated fusion + output head ----
  launch_gemm(stream, Wcat, 256, g1_W, 128, g1_b, W0, 128, NN, 256, 1.0f, GF_GELU);
  gate_kernel<<<NB8, 256, 0, stream>>>(W0, g2_W, g2_b, Gt, NN);
  fuse_kernel<<<NBn, 256, 0, stream>>>(Wcat, Gt, W1, NHL);
  launch_gemm(stream, W1, 128, o1_W, 128, o1_b, W2, 128, NN, 128, 1.0f, GF_GELU);
  launch_gemm(stream, W2, 128, o2_W, 128, o2_b, (float*)d_out, 128, NN, 128, 1.0f, 0);
}
